// IntegralTransform_79886391705729
// MI455X (gfx1250) — matrix operand order repaired and checked by reading, not compiled, not run on hardware
//
#include <hip/hip_runtime.h>
#include <math.h>

// IntegralTransform (graph neural operator), MI455X / gfx1250.
// Per-edge MLP 38->64->64->32 via v_wmma_f32_16x16x32_f16; each wave32
// processes one point's K=32 edges as two 16-row WMMA tiles. MLP weights are
// staged to LDS with the Tensor Data Mover (tensor_load_to_lds + tensorcnt),
// converted once to f16 column-major. Segment-sum is a pure in-wave reduce.

typedef __attribute__((ext_vector_type(16))) _Float16 v16h;
typedef __attribute__((ext_vector_type(8)))  _Float16 v8h;
typedef __attribute__((ext_vector_type(2)))  __fp16   v2hraw;   // cvt_pkrtz result type
typedef __attribute__((ext_vector_type(8)))  float    v8f;
typedef __attribute__((ext_vector_type(4)))  unsigned int u32x4;
typedef __attribute__((ext_vector_type(8)))  int      i32x8;
typedef __attribute__((ext_vector_type(4)))  int      i32x4;

#define WAVES_PER_BLOCK 8
#define TILE_M 16
#define HID 64          // hidden width; padded-K stride for all LDS tiles
#define CCH 32          // channels C
#define DIM 3
#define IN_FEATS (2 * DIM + CCH)   // 38, zero-padded to 64

#define W1_ELEMS (IN_FEATS * HID)  // 2432 f32
#define W2_ELEMS (HID * HID)       // 4096 f32
#define W3_ELEMS (HID * CCH)       // 2048 f32
#define STAGE_ELEMS (W1_ELEMS + W2_ELEMS + W3_ELEMS)

#if __has_builtin(__builtin_amdgcn_tensor_load_to_lds) && \
    __has_builtin(__builtin_amdgcn_s_wait_tensorcnt)
#define HAVE_TDM 1
#else
#define HAVE_TDM 0
#endif

// cheap gelu: x * sigmoid(1.702 x)  (v_exp_f32 + v_rcp_f32, ~5 VALU)
__device__ __forceinline__ float gelu_fast(float x) {
    const float s = __builtin_amdgcn_rcpf(1.0f + __expf(-1.702f * x));
    return x * s;
}

// A-fragment (16x32 f16, MxK): lane l holds row (l&15);
// half-wave 0 holds K = kt*32 + {0..15}, half-wave 1 K = kt*32 + {16..31}.
__device__ __forceinline__ v16h load_a_frag(const _Float16* tile, int lane, int kt) {
    const int r = lane & 15;
    const int h = lane >> 4;
    const _Float16* p = tile + r * HID + kt * 32 + h * 16;
    return *(const v16h*)(p);
}

// B-fragment (32x16 f16, KxN): lane l holds column n = 16*j + (l&15);
// 16 contiguous halves when weights are column-major [n][k], k-stride 64.
__device__ __forceinline__ v16h load_b_frag(const _Float16* wcolmaj, int lane, int j, int kt) {
    const int n = (j << 4) + (lane & 15);
    const int h = lane >> 4;
    return *(const v16h*)(wcolmaj + n * HID + kt * 32 + h * 16);
}

__device__ __forceinline__ v8f wmma_f16(v16h a, v16h b, v8f c) {
    return __builtin_amdgcn_wmma_f32_16x16x32_f16(false, a, false, b,
                                                  (short)0, c, false, false);
}

#if HAVE_TDM
// Issue a 1-D TDM copy of nelem f32 from global to LDS (D# per ISA ch.8).
// 6-arg builtin form: (u32x4 g0, i32x8 g1, i32x4, i32x4, i32x8, i32 cpol).
__device__ __forceinline__ void tdm_load_1d_f32(const float* gsrc,
                                                unsigned lds_byte_off, int nelem) {
    const unsigned long long ga = (unsigned long long)(uintptr_t)gsrc;
    u32x4 g0;
    g0[0] = 1u;                                     // count=1, user D#
    g0[1] = lds_byte_off;                           // lds_addr (bytes)
    g0[2] = (unsigned)(ga & 0xFFFFFFFFu);           // global_addr[31:0]
    g0[3] = (unsigned)((ga >> 32) & 0x1FFFFFFu)     // global_addr[56:32]
          | (2u << 30);                             // type=2 ("image")
    i32x8 g1;
    g1[0] = (2 << 16);                              // data_size=2 -> 4 bytes
    g1[1] = (nelem & 0xFFFF) << 16;                 // tensor_dim0[15:0]
    g1[2] = (nelem >> 16) & 0xFFFF;                 // tensor_dim0[31:16]
    g1[3] = (nelem & 0xFFFF) << 16;                 // tile_dim0 (<=65535)
    g1[4] = 0;                                      // tile_dim1/2 unused
    g1[5] = nelem;                                  // tensor_dim0_stride[31:0]
    g1[6] = 0;
    g1[7] = 0;
    const i32x4 z4 = {0, 0, 0, 0};                  // groups 2/3 disabled
    const i32x8 z8 = {0, 0, 0, 0, 0, 0, 0, 0};
    __builtin_amdgcn_tensor_load_to_lds(g0, g1, z4, z4, z8, 0);
}
#endif

__global__ void __launch_bounds__(256, 1)
integral_transform_kernel(const float* __restrict__ y,
                          const float* __restrict__ f_y,
                          const float* __restrict__ wts,
                          const float* __restrict__ W1, const float* __restrict__ b1,
                          const float* __restrict__ W2, const float* __restrict__ b2,
                          const float* __restrict__ W3, const float* __restrict__ b3,
                          const int*   __restrict__ nidx,
                          const int*   __restrict__ rowsplits,
                          float* __restrict__ out,
                          int N, int K) {
    // f16 weights, column-major [n][k], k padded to 64
    __shared__ __align__(32) _Float16 sW1[HID * HID];
    __shared__ __align__(32) _Float16 sW2[HID * HID];
    __shared__ __align__(32) _Float16 sW3[CCH * HID];
    __shared__ float sB1[HID], sB2[HID], sB3[CCH];
    // per-wave tiles
    __shared__ __align__(32) _Float16 sA[WAVES_PER_BLOCK][TILE_M * HID];
    __shared__ __align__(32) _Float16 sH[WAVES_PER_BLOCK][TILE_M * HID];
    __shared__ float sWv[WAVES_PER_BLOCK][TILE_M];
    // raw f32 staging area for TDM weight DMA
    __shared__ __align__(16) float sStage[STAGE_ELEMS];

    const int tid = threadIdx.x;

    // ---- weight staging: TDM DMA (global -> LDS), then convert to f16 ----
    const float* W1src = W1;
    const float* W2src = W2;
    const float* W3src = W3;
#if HAVE_TDM
    if (tid < 32) {   // one wave issues the DMAs (EXEC ignored by TDM)
        const unsigned base = (unsigned)(uintptr_t)&sStage[0];
        tdm_load_1d_f32(W1, base, W1_ELEMS);
        tdm_load_1d_f32(W2, base + W1_ELEMS * 4u, W2_ELEMS);
        tdm_load_1d_f32(W3, base + (W1_ELEMS + W2_ELEMS) * 4u, W3_ELEMS);
        __builtin_amdgcn_s_wait_tensorcnt(0);
    }
    W1src = &sStage[0];
    W2src = &sStage[W1_ELEMS];
    W3src = &sStage[W1_ELEMS + W2_ELEMS];
#endif
    __syncthreads();

    for (int i = tid; i < HID * HID; i += blockDim.x) {
        const int n = i / HID, k = i % HID;
        sW1[i] = (_Float16)((k < IN_FEATS) ? W1src[k * HID + n] : 0.0f);
        sW2[i] = (_Float16)W2src[k * HID + n];
    }
    for (int i = tid; i < CCH * HID; i += blockDim.x) {
        const int n = i / HID, k = i % HID;
        sW3[i] = (_Float16)W3src[k * CCH + n];
    }
    if (tid < HID) { sB1[tid] = b1[tid]; sB2[tid] = b2[tid]; }
    if (tid < CCH) { sB3[tid] = b3[tid]; }
    __syncthreads();

    const int wave  = tid >> 5;
    const int lane  = tid & 31;
    const int r     = lane & 15;
    const int hl    = lane >> 4;
    const int nsub  = lane & 15;
    _Float16* Atile = &sA[wave][0];
    _Float16* Htile = &sH[wave][0];
    float*    Wv    = &sWv[wave][0];

    const int gwave  = blockIdx.x * WAVES_PER_BLOCK + wave;
    const int nwaves = gridDim.x * WAVES_PER_BLOCK;
    const int ntiles = K >> 4;          // K = 32 here (uniform CSR)

    for (int p = gwave; p < N; p += nwaves) {
        const int e0 = rowsplits[p];
        if (p + nwaves < N)
            __builtin_prefetch(&nidx[(long)(p + nwaves) * K], 0, 1);  // global_prefetch_b8

        float acc_out0 = 0.0f, acc_out1 = 0.0f;

        for (int t = 0; t < ntiles; ++t) {
            // ---- stage 16-edge tile: [y[nbr], y[p], f_y[nbr]] pad-> 64 f16 ----
            const int e  = e0 + t * TILE_M + r;
            const int nb = nidx[e];
            _Float16* Arow = Atile + r * HID;
            const float4* fr = (const float4*)&f_y[(long)nb * CCH];
            if (hl == 0) {
                // coords, packed f16 pairs (v_cvt_pk_f16_f32 -> ds_store_b32)
                *(v2hraw*)&Arow[0] = __builtin_amdgcn_cvt_pkrtz(y[nb * DIM + 0], y[nb * DIM + 1]);
                *(v2hraw*)&Arow[2] = __builtin_amdgcn_cvt_pkrtz(y[nb * DIM + 2], y[(long)p * DIM + 0]);
                *(v2hraw*)&Arow[4] = __builtin_amdgcn_cvt_pkrtz(y[(long)p * DIM + 1], y[(long)p * DIM + 2]);
                #pragma unroll
                for (int q = 0; q < 4; ++q) {             // f_y channels 0..15
                    const float4 v = fr[q];
                    *(v2hraw*)&Arow[6 + 4 * q + 0] = __builtin_amdgcn_cvt_pkrtz(v.x, v.y);
                    *(v2hraw*)&Arow[6 + 4 * q + 2] = __builtin_amdgcn_cvt_pkrtz(v.z, v.w);
                }
                Wv[r] = wts[nb];
            } else {
                #pragma unroll
                for (int q = 4; q < 8; ++q) {             // f_y channels 16..31
                    const float4 v = fr[q];
                    *(v2hraw*)&Arow[6 + 4 * q + 0] = __builtin_amdgcn_cvt_pkrtz(v.x, v.y);
                    *(v2hraw*)&Arow[6 + 4 * q + 2] = __builtin_amdgcn_cvt_pkrtz(v.z, v.w);
                }
                #pragma unroll
                for (int q = 0; q < (HID - IN_FEATS) / 2; ++q)  // zero pad 38..63
                    *(unsigned int*)&Arow[IN_FEATS + 2 * q] = 0u;
            }

            // ---- layer 1: [16x64pad] @ W1[64x64] + b1, gelu -> Htile ----
            v16h a0 = load_a_frag(Atile, lane, 0);
            v16h a1 = load_a_frag(Atile, lane, 1);
            #pragma unroll
            for (int j = 0; j < 4; ++j) {
                v8f acc = {};
                acc = wmma_f16(a0, load_b_frag(sW1, lane, j, 0), acc);
                acc = wmma_f16(a1, load_b_frag(sW1, lane, j, 1), acc);
                const float bias = sB1[(j << 4) + nsub];
                #pragma unroll
                for (int v = 0; v < 8; ++v) {
                    const float x = gelu_fast(acc[v] + bias);
                    Htile[(v + 8 * hl) * HID + (j << 4) + nsub] = (_Float16)x;
                }
            }

            // ---- layer 2: h1[16x64] @ W2[64x64] + b2, gelu -> Htile ----
            a0 = load_a_frag(Htile, lane, 0);
            a1 = load_a_frag(Htile, lane, 1);
            v8f h2acc[4];
            #pragma unroll
            for (int j = 0; j < 4; ++j) {
                v8f acc = {};
                acc = wmma_f16(a0, load_b_frag(sW2, lane, j, 0), acc);
                acc = wmma_f16(a1, load_b_frag(sW2, lane, j, 1), acc);
                h2acc[j] = acc;
            }
            #pragma unroll
            for (int j = 0; j < 4; ++j) {
                const float bias = sB2[(j << 4) + nsub];
                #pragma unroll
                for (int v = 0; v < 8; ++v) {
                    const float x = gelu_fast(h2acc[j][v] + bias);
                    Htile[(v + 8 * hl) * HID + (j << 4) + nsub] = (_Float16)x;
                }
            }

            // ---- layer 3: h2[16x64] @ W3[64x32] + b3; *f_y *w; row-reduce ----
            a0 = load_a_frag(Htile, lane, 0);
            a1 = load_a_frag(Htile, lane, 1);
            #pragma unroll
            for (int j = 0; j < 2; ++j) {
                v8f acc = {};
                acc = wmma_f16(a0, load_b_frag(sW3, lane, j, 0), acc);
                acc = wmma_f16(a1, load_b_frag(sW3, lane, j, 1), acc);
                const float bias = sB3[(j << 4) + nsub];
                float psum = 0.0f;
                #pragma unroll
                for (int v = 0; v < 8; ++v) {
                    const int M = v + 8 * hl;
                    const float fy = (float)Atile[M * HID + 6 + (j << 4) + nsub];
                    psum += (acc[v] + bias) * fy * Wv[M];
                }
                if (j == 0) acc_out0 += psum; else acc_out1 += psum;
            }
        }

        // fold half-wave row subsets; lane then has full sum for its channel
        acc_out0 += __shfl_xor(acc_out0, 16, 32);
        acc_out1 += __shfl_xor(acc_out1, 16, 32);
        const float vout = (lane < 16) ? acc_out0 : acc_out1;
        out[(long)p * CCH + lane] = vout;
    }
}

extern "C" void kernel_launch(void* const* d_in, const int* in_sizes, int n_in,
                              void* d_out, int out_size, void* d_ws, size_t ws_size,
                              hipStream_t stream) {
    const float* y   = (const float*)d_in[0];
    const float* f_y = (const float*)d_in[1];
    const float* w   = (const float*)d_in[2];
    const float* W1  = (const float*)d_in[3];
    const float* b1  = (const float*)d_in[4];
    const float* W2  = (const float*)d_in[5];
    const float* b2  = (const float*)d_in[6];
    const float* W3  = (const float*)d_in[7];
    const float* b3  = (const float*)d_in[8];
    const int* nidx  = (const int*)d_in[9];
    const int* rs    = (const int*)d_in[10];
    float* out = (float*)d_out;

    const int N = in_sizes[2];          // weights is [N,1]
    const int E = in_sizes[9];          // neighbors_index
    const int K = E / N;                // uniform CSR (harness: 32)

    const int blocks = (N + WAVES_PER_BLOCK - 1) / WAVES_PER_BLOCK;
    integral_transform_kernel<<<blocks, 256, 0, stream>>>(
        y, f_y, w, W1, b1, W2, b2, W3, b3, nidx, rs, out, N, K);
}
